// LinearAttention_30820685316749
// MI455X (gfx1250) — matrix operand order repaired and checked by reading, not compiled, not run on hardware
//
#include <hip/hip_runtime.h>
#include <hip/hip_fp16.h>

typedef __attribute__((ext_vector_type(16))) _Float16 v16h;
typedef __attribute__((ext_vector_type(8)))  _Float16 v8h;
typedef __attribute__((ext_vector_type(8)))  float    v8f;
typedef __attribute__((ext_vector_type(4)))  float    v4f;

#define N_    4
#define L_    8192
#define H_    8
#define D_    128
#define SPLIT 16
#define CHUNK (L_ / SPLIT)     // 512 kv positions per workgroup in pass 1
#define LBLK  (L_ / 128)       // 64 row-blocks per (n,h) in pass 2
#define EPS_  1e-6f

// phi(x) = elu(x) + 1  (strictly positive feature map)
__device__ __forceinline__ float phi(float x) { return x > 0.f ? x + 1.f : __expf(x); }

__device__ __forceinline__ v8f wmma_f16(const v16h& a, const v16h& b, const v8f& c) {
    return __builtin_amdgcn_wmma_f32_16x16x32_f16(false, a, false, b, (short)0, c, false, false);
}

// ---------------------------------------------------------------------------
// Workspace zeroing (harness poisons d_ws; pass 1 accumulates with atomics)
// ---------------------------------------------------------------------------
__global__ void la_zero(float* __restrict__ p, int count) {
    int i = blockIdx.x * blockDim.x + threadIdx.x;
    if (i < count) p[i] = 0.f;
}

// ---------------------------------------------------------------------------
// Pass 1: S1[d,v] = sum_s phi(K[s,d]) * V[s,v]   and   ksum[d] = sum_s phi(K[s,d])
// grid = N*H*SPLIT workgroups of 256 threads (8 waves).
// Wave w owns d-rows [16w,16w+16) x all 128 v columns (8 accumulator tiles).
// ---------------------------------------------------------------------------
__global__ __launch_bounds__(256) void la_kv(const float* __restrict__ Kg,
                                             const float* __restrict__ Vg,
                                             float* __restrict__ wsKV,
                                             float* __restrict__ wsKs) {
    __shared__ __align__(16) _Float16 ldsKT[128 * 40]; // [d][s], padded to 40 halves
    __shared__ __align__(16) _Float16 ldsVT[128 * 40];
    __shared__ float ldsR[256];

    const int wg    = blockIdx.x;
    const int chunk = wg % SPLIT;
    const int nh    = wg / SPLIT;
    const int h     = nh % H_;
    const int n     = nh / H_;
    const int t     = threadIdx.x;
    const int lane  = t & 31;
    const int wave  = t >> 5;

    const size_t base_nh = ((size_t)n * L_) * (H_ * D_) + (size_t)h * D_;
    const int s_begin = chunk * CHUNK;

    // cooperative-load mapping: element e = t + 256*i -> s = (t>>7)+2i, d = t&127
    const int dcol  = t & 127;
    const int srow0 = t >> 7;

    v8f acc[8];
#pragma unroll
    for (int i = 0; i < 8; i++)
#pragma unroll
        for (int j = 0; j < 8; j++) acc[i][j] = 0.f;

    float ksumPart = 0.f;

    for (int s0 = s_begin; s0 < s_begin + CHUNK; s0 += 32) {
        __syncthreads();
        // stage 32 x 128 slabs of phi(K) and V into LDS, transposed to [d][s]
#pragma unroll
        for (int i = 0; i < 16; i++) {
            int s_local = srow0 + 2 * i;
            size_t gidx = base_nh + (size_t)(s0 + s_local) * (H_ * D_) + dcol;
            float pk = phi(Kg[gidx]);
            ksumPart += pk;
            ldsKT[dcol * 40 + s_local] = (_Float16)pk;
            ldsVT[dcol * 40 + s_local] = (_Float16)Vg[gidx];
        }
        __syncthreads();

        // A fragment: 16(d) x 32(s).  lane = m + 16g; halves 0-7 <-> k=g*8+0..7,
        // halves 8-15 <-> k=16+g*8+0..7 (contiguous 16B runs in ldsKT).
        {
            const int m = lane & 15;
            const int g = lane >> 4;
            const int drow = wave * 16 + m;
            const v8h* rowK = (const v8h*)(&ldsKT[drow * 40]);
            v8h a_lo = rowK[g];
            v8h a_hi = rowK[2 + g];
            v16h afrag;
#pragma unroll
            for (int j = 0; j < 8; j++) { afrag[j] = a_lo[j]; afrag[8 + j] = a_hi[j]; }

#pragma unroll
            for (int ct = 0; ct < 8; ct++) {
                // B fragment: 32(s) x 16(v). lane = n + 16g holds k = g*8 + 0..7, 16 + g*8 + 0..7.
                const int vcol = ct * 16 + m;
                const v8h* rowV = (const v8h*)(&ldsVT[vcol * 40]);
                v8h b_lo = rowV[g];
                v8h b_hi = rowV[2 + g];
                v16h bfrag;
#pragma unroll
                for (int j = 0; j < 8; j++) { bfrag[j] = b_lo[j]; bfrag[8 + j] = b_hi[j]; }
                acc[ct] = wmma_f16(afrag, bfrag, acc[ct]);
            }
        }
    }

    // ksum reduction: threads t and t+128 share d = t&127
    ldsR[t] = ksumPart;
    __syncthreads();
    if (t < 128) {
        float s = ldsR[t] + ldsR[t + 128];
        atomicAdd(&wsKs[nh * 128 + t], s);
    }

    // merge partial 128x128 tile.  C layout: lanes 0-15: M=r, lanes 16-31: M=8+r; N=lane&15
    {
        const int m  = lane & 15;
        const int hi = lane >> 4;
        float* kvbase = wsKV + (size_t)nh * (128 * 128);
#pragma unroll
        for (int ct = 0; ct < 8; ct++)
#pragma unroll
            for (int r = 0; r < 8; r++) {
                int dd = wave * 16 + r + 8 * hi;
                int vv = ct * 16 + m;
                atomicAdd(&kvbase[dd * 128 + vv], acc[ct][r]);
            }
    }
}

// ---------------------------------------------------------------------------
// Pass 2: out[l,v] = (phiQ[l,:] @ S1[:,v]) / (phiQ[l,:] . ksum + eps)
// grid = N*H*(L/128); each WG caches S1 (f16, transposed) and handles 128 rows.
// ---------------------------------------------------------------------------
__global__ __launch_bounds__(256) void la_out(const float* __restrict__ Qg,
                                              const float* __restrict__ wsKV,
                                              const float* __restrict__ wsKs,
                                              float* __restrict__ outg) {
    __shared__ __align__(16) _Float16 ldsKVT[128 * 136]; // [v][d], padded rows
    __shared__ float ldsKs[128];

    const int wg   = blockIdx.x;
    const int lblk = wg % LBLK;
    const int nh   = wg / LBLK;
    const int h    = nh % H_;
    const int n    = nh / H_;
    const int t    = threadIdx.x;
    const int lane = t & 31;
    const int wave = t >> 5;

    // stage S1 into LDS transposed [v][d] as f16 (coalesced global reads)
    {
        const float* kvbase = wsKV + (size_t)nh * (128 * 128);
        const int vcol = t & 127;
        const int dr0  = t >> 7;
#pragma unroll
        for (int i = 0; i < 64; i++) {
            int dd = dr0 + 2 * i;
            ldsKVT[vcol * 136 + dd] = (_Float16)kvbase[dd * 128 + vcol];
        }
        if (t < 128) ldsKs[t] = wsKs[nh * 128 + t];
    }
    __syncthreads();

    const int m = lane & 15;
    const int g = lane >> 4;
    const int l = lblk * 128 + wave * 16 + m;
    const float* qrow = Qg + ((size_t)(n * L_ + l) * H_ + h) * D_;

    // A fragments over d (4 k-steps of 32), phi in fp32, z in fp32
    v16h afrag[4];
    float zpart = 0.f;
#pragma unroll
    for (int kk = 0; kk < 4; kk++) {
        const int d0 = kk * 32 + g * 8;
        const int d1 = kk * 32 + 16 + g * 8;
        v4f qa0 = *(const v4f*)(qrow + d0);
        v4f qa1 = *(const v4f*)(qrow + d0 + 4);
        v4f qb0 = *(const v4f*)(qrow + d1);
        v4f qb1 = *(const v4f*)(qrow + d1 + 4);
#pragma unroll
        for (int j = 0; j < 4; j++) {
            float p0 = phi(qa0[j]); zpart += p0 * ldsKs[d0 + j];     afrag[kk][j]      = (_Float16)p0;
            float p1 = phi(qa1[j]); zpart += p1 * ldsKs[d0 + 4 + j]; afrag[kk][4 + j]  = (_Float16)p1;
            float p2 = phi(qb0[j]); zpart += p2 * ldsKs[d1 + j];     afrag[kk][8 + j]  = (_Float16)p2;
            float p3 = phi(qb1[j]); zpart += p3 * ldsKs[d1 + 4 + j]; afrag[kk][12 + j] = (_Float16)p3;
        }
    }
    // lane m and lane m+16 hold complementary halves of the d-sum
    float z = zpart + __shfl_xor(zpart, 16, 32);
    float zinv = 1.f / (z + EPS_);

    v8f acc[8];
#pragma unroll
    for (int i = 0; i < 8; i++)
#pragma unroll
        for (int j = 0; j < 8; j++) acc[i][j] = 0.f;

#pragma unroll
    for (int ct = 0; ct < 8; ct++) {
#pragma unroll
        for (int kk = 0; kk < 4; kk++) {
            const v8h* row = (const v8h*)(&ldsKVT[(ct * 16 + m) * 136 + kk * 32]);
            v8h b_lo = row[g];
            v8h b_hi = row[2 + g];
            v16h bfrag;
#pragma unroll
            for (int j = 0; j < 8; j++) { bfrag[j] = b_lo[j]; bfrag[8 + j] = b_hi[j]; }
            acc[ct] = wmma_f16(afrag[kk], bfrag, acc[ct]);
        }
    }

    // scale by zinv of row (r + 8g) and store to merged-head output [N, L, H*128]
#pragma unroll
    for (int r = 0; r < 8; r++) {
        float zr = __shfl(zinv, g * 8 + r, 32);
        int lr = lblk * 128 + wave * 16 + 8 * g + r;
        size_t rowbase = ((size_t)(n * L_) + lr) * (H_ * D_) + h * D_;
#pragma unroll
        for (int ct = 0; ct < 8; ct++) {
            outg[rowbase + ct * 16 + m] = acc[ct][r] * zr;
        }
    }
}

// ---------------------------------------------------------------------------
extern "C" void kernel_launch(void* const* d_in, const int* in_sizes, int n_in,
                              void* d_out, int out_size, void* d_ws, size_t ws_size,
                              hipStream_t stream) {
    (void)in_sizes; (void)n_in; (void)out_size; (void)ws_size;
    const float* Q = (const float*)d_in[0];
    const float* K = (const float*)d_in[1];
    const float* V = (const float*)d_in[2];
    float* out = (float*)d_out;

    float* wsKV = (float*)d_ws;                              // N*H*128*128 floats
    float* wsKs = wsKV + (size_t)N_ * H_ * 128 * 128;        // N*H*128 floats

    int zcount = N_ * H_ * (128 * 128 + 128);
    la_zero<<<(zcount + 255) / 256, 256, 0, stream>>>(wsKV, zcount);
    la_kv<<<N_ * H_ * SPLIT, 256, 0, stream>>>(K, V, wsKV, wsKs);
    la_out<<<N_ * H_ * LBLK, 256, 0, stream>>>(Q, wsKV, wsKs, out);
}
